// CGConv_by_occurrence_global_pool_19121194402168
// MI455X (gfx1250) — compile-verified, matrix operand order checked by reading, not run on hardware
//
#include <hip/hip_runtime.h>
#include <hip/hip_bf16.h>
#include <algorithm>

// ---------------- types for WMMA ----------------
typedef __attribute__((ext_vector_type(16))) __bf16    v16bf;
typedef __attribute__((ext_vector_type(8)))  float     v8f;
typedef __attribute__((ext_vector_type(8)))  unsigned  v8u;

#define C_DIM 64
#define ZIN   132      // 2*C + 4
#define KPAD  160      // 5 k-steps of 32
#define NLAYERS 3
#define FRAGS_PER_LAYER 40          // 2 mats * 5 ksteps * 4 ntiles
#define FRAG_ELEMS 512              // 32 lanes * 16 bf16
#define BPACK_L_ELEMS (FRAGS_PER_LAYER*FRAG_ELEMS)   // 20480 ushorts per layer

__device__ __forceinline__ unsigned bf16bits(float x){
    unsigned u = __float_as_uint(x);
    return (u + 0x7FFFu + ((u >> 16) & 1u)) >> 16;     // RNE
}
__device__ __forceinline__ unsigned packbf(float x, float y){
    return (bf16bits(y) << 16) | (bf16bits(x) & 0xFFFFu);
}

// ---------------- weight packing into WMMA-B fragment layout ----------------
// Bpack element index = ((l*2+mat)*5+ks)*4+nt)*512 + lane*16 + j
// B element for lane,j : N = nt*16 + (lane&15); kb = lane<16 ? 0 : 8;
//   d=j>>1 ; K = ks*32 + (d<4 ? kb + j : 16 + kb + (j-8))   (pairs of consecutive K)
__global__ void pack_weights(const float* __restrict__ Wf, const float* __restrict__ Ws,
                             unsigned short* __restrict__ Bpack){
    int idx = blockIdx.x*256 + threadIdx.x;
    if (idx >= NLAYERS*FRAGS_PER_LAYER*FRAG_ELEMS) return;
    int j    = idx & 15;
    int lane = (idx >> 4) & 31;
    int frag = idx >> 9;                 // 0..119
    int nt = frag & 3;
    int t  = frag >> 2;                  // 0..29
    int ks = t % 5;
    int u  = t / 5;                      // 0..5
    int mat = u & 1;
    int l   = u >> 1;
    int N  = nt*16 + (lane & 15);
    int kb = (lane < 16) ? 0 : 8;
    int d  = j >> 1;
    int K  = ks*32 + (d < 4 ? kb + j : 16 + kb + (j - 8));
    const float* W = mat ? Ws : Wf;      // [L][C][ZIN] row-major, B[k][n] = W[n][k]
    float v = (K < ZIN) ? W[((size_t)l*C_DIM + N)*ZIN + K] : 0.0f;
    Bpack[idx] = (unsigned short)bf16bits(v);
}

// ---------------- small helper kernels ----------------
__global__ void fill_f32(float* p, float v, int n){
    int i = blockIdx.x*256 + threadIdx.x; if (i < n) p[i] = v;
}
__global__ void fill_u32(unsigned* p, unsigned v, int n){
    int i = blockIdx.x*256 + threadIdx.x; if (i < n) p[i] = v;
}
__global__ void copy_x(const float* __restrict__ x, float* __restrict__ h, int n){
    int i = blockIdx.x*256 + threadIdx.x; if (i < n) h[i] = x[i];
}
__global__ void prep_ea(const float* __restrict__ ea, float* __restrict__ ea4, int E){
    int e = blockIdx.x*256 + threadIdx.x;
    if (e >= E) return;
    const float* r = ea + (size_t)e*25;
    float* o = ea4 + (size_t)e*4;
    o[0] = r[6]  * 0.2f;  o[1] = r[9]  * 0.2f;
    o[2] = r[21] * 0.2f;  o[3] = r[24] * 0.2f;
}

// ---------------- WMMA edge-message kernel (one layer) ----------------
__global__ __launch_bounds__(128) void edge_msg(
    const float* __restrict__ h, const int* __restrict__ srcp, const int* __restrict__ dstp,
    const float* __restrict__ ea4, const unsigned short* __restrict__ bpackL,
    const float* __restrict__ bfv, const float* __restrict__ bsv,
    float* __restrict__ agg, int E, int numTiles)
{
    __shared__ __align__(32) unsigned blds[FRAGS_PER_LAYER*256]; // 40 KB, dwords
    __shared__ __align__(16) unsigned zlds[4*16*(KPAD/2)];       // 20 KB (4 waves x 16 rows x 80 dw)
    __shared__ int dstl[64];

    // stage this layer's packed B fragments in LDS once per block
    const unsigned* bsrc = (const unsigned*)bpackL;
    for (int i = threadIdx.x; i < FRAGS_PER_LAYER*256; i += 128) blds[i] = bsrc[i];
    __syncthreads();

    const int Lid  = threadIdx.x & 31;
    const int wave = threadIdx.x >> 5;
    const int row  = Lid & 15;
    const int half = Lid >> 4;
    const int kb   = half ? 8 : 0;
    const v8f zerov = {0.f,0.f,0.f,0.f,0.f,0.f,0.f,0.f};

    for (int tile = blockIdx.x*4 + wave; tile < numTiles; tile += gridDim.x*4) {
        const int tb = tile*16;
        const int e  = tb + row;
        const int ec = (e < E) ? e : (E - 1);
        const unsigned zb = (unsigned)(wave*16 + row)*(KPAD/2);

        // ---- build bf16 z-row (16 edges x 160) in LDS ----
        if (half == 0) {                       // K 0..63 = h[dst], pad dwords 66..72
            int dv = dstp[ec];
            dstl[wave*16 + row] = dv;
            const float4* hp = (const float4*)(h + (size_t)dv*C_DIM);
            #pragma unroll
            for (int i = 0; i < 16; ++i) {
                float4 t = hp[i];
                zlds[zb + 2*i]     = packbf(t.x, t.y);
                zlds[zb + 2*i + 1] = packbf(t.z, t.w);
            }
            #pragma unroll
            for (int i = 66; i < 73; ++i) zlds[zb + i] = 0u;
        } else {                               // K 64..131 = h[src] + ea, pad 73..79
            int sv = srcp[ec];
            const float4* hp = (const float4*)(h + (size_t)sv*C_DIM);
            #pragma unroll
            for (int i = 0; i < 16; ++i) {
                float4 t = hp[i];
                zlds[zb + 32 + 2*i]     = packbf(t.x, t.y);
                zlds[zb + 32 + 2*i + 1] = packbf(t.z, t.w);
            }
            float4 ev = *(const float4*)(ea4 + (size_t)ec*4);
            zlds[zb + 64] = packbf(ev.x, ev.y);
            zlds[zb + 65] = packbf(ev.z, ev.w);
            #pragma unroll
            for (int i = 73; i < 80; ++i) zlds[zb + i] = 0u;
        }
        __builtin_amdgcn_wave_barrier();       // DS in-order within wave32 -> data visible

        // ---- 40 x v_wmma_f32_16x16x32_bf16 ----
        v8f accf[4] = {zerov, zerov, zerov, zerov};
        v8f accs[4] = {zerov, zerov, zerov, zerov};
        const unsigned rowdw = zb;             // A lane L -> row (L&15), same as z builder
        #pragma unroll
        for (int ks = 0; ks < 5; ++ks) {
            v8u au;
            #pragma unroll
            for (int d = 0; d < 8; ++d) {      // 16-bit A layout: pairs of consecutive K
                int K0 = ks*32 + (d < 4 ? kb + 2*d : 16 + kb + 2*(d - 4));
                au[d] = zlds[rowdw + (K0 >> 1)];
            }
            v16bf a = __builtin_bit_cast(v16bf, au);
            #pragma unroll
            for (int nt = 0; nt < 4; ++nt) {
                v8u bu  = *(const v8u*)&blds[((0*5 + ks)*4 + nt)*256 + Lid*8];
                v16bf b = __builtin_bit_cast(v16bf, bu);
                accf[nt] = __builtin_amdgcn_wmma_f32_16x16x32_bf16(
                    false, a, false, b, (short)0, accf[nt], false, false);
                v8u bu2  = *(const v8u*)&blds[((1*5 + ks)*4 + nt)*256 + Lid*8];
                v16bf b2 = __builtin_bit_cast(v16bf, bu2);
                accs[nt] = __builtin_amdgcn_wmma_f32_16x16x32_bf16(
                    false, a, false, b2, (short)0, accs[nt], false, false);
            }
        }
        __builtin_amdgcn_wave_barrier();

        // ---- gate + scatter-add: C/D layout lane L holds N=L&15, M = r + 8*(L>>4) ----
        #pragma unroll
        for (int nt = 0; nt < 4; ++nt) {
            const int ch = nt*16 + row;
            const float bfc = bfv[ch], bsc = bsv[ch];
            #pragma unroll
            for (int r = 0; r < 8; ++r) {
                const int rr = r + half*8;
                const int e2 = tb + rr;
                if (e2 < E) {
                    float f = accf[nt][r] + bfc;
                    float s = accs[nt][r] + bsc;
                    float sig = 1.0f / (1.0f + __expf(-f));
                    float sp  = (s > 20.0f) ? s : log1pf(__expf(s));
                    int dn = dstl[wave*16 + rr];
                    atomicAdd(agg + (size_t)dn*C_DIM + ch, sig * sp);
                }
            }
        }
    }
}

// ---------------- BN(eval) + residual + tanh, write h and JK slice ----------------
__global__ void node_update(const float* __restrict__ agg, float* __restrict__ h,
                            float* __restrict__ xcat,
                            const float* __restrict__ g, const float* __restrict__ b,
                            const float* __restrict__ m, const float* __restrict__ v,
                            int l, int n64){
    int i = blockIdx.x*256 + threadIdx.x;
    if (i >= n64) return;
    int c = i & 63, n = i >> 6;
    float a  = (agg[i] - m[c]) * rsqrtf(v[c] + 1e-5f) * g[c] + b[c];
    float hn = tanhf(a + h[i]);
    h[i] = hn;
    xcat[(size_t)n*(C_DIM*NLAYERS) + l*C_DIM + c] = hn;
}

// ---------------- interface pooling (add / count / max via ordered keys) -------
__global__ void pool_scatter(const float* __restrict__ xcat, const int* __restrict__ ipos,
                             const int* __restrict__ batch, float* __restrict__ padd,
                             float* __restrict__ pcnt, unsigned* __restrict__ pmax, int Ni){
    int idx = blockIdx.x*256 + threadIdx.x;
    if (idx >= Ni*192) return;
    int i = idx / 192, d = idx % 192;
    int node = ipos[i];
    int g = batch[node];
    float v = xcat[(size_t)node*192 + d];
    atomicAdd(padd + g*192 + d, v);
    unsigned uv = __float_as_uint(v);
    unsigned key = (uv & 0x80000000u) ? ~uv : (uv | 0x80000000u);
    atomicMax(pmax + g*192 + d, key);
    if (d == 0) atomicAdd(pcnt + g, 1.0f);
}

// ---------------- final MLP: one block (288 thr) per graph ----------------
__global__ void mlp_kernel(const float* __restrict__ padd, const float* __restrict__ pcnt,
                           const unsigned* __restrict__ pmax,
                           const float* __restrict__ W1, const float* __restrict__ b1,
                           const float* __restrict__ W2, const float* __restrict__ b2,
                           float* __restrict__ out){
    int g = blockIdx.x, t = threadIdx.x;           // 288 threads
    __shared__ float pooled[576];
    __shared__ float red[288];
    for (int d = t; d < 576; d += 288) {
        float v;
        if (d < 192)       v = padd[g*192 + d];
        else if (d < 384)  v = padd[g*192 + (d - 192)] / fmaxf(pcnt[g], 1.0f);
        else {
            unsigned k = pmax[g*192 + (d - 384)];
            unsigned bits = (k & 0x80000000u) ? (k & 0x7FFFFFFFu) : ~k;
            v = __uint_as_float(bits);
        }
        pooled[d] = v;
    }
    __syncthreads();
    float acc = b1[t];
    for (int d = 0; d < 576; ++d) acc += pooled[d] * W1[d*288 + t];
    red[t] = fmaxf(acc, 0.0f) * W2[t];
    __syncthreads();
    if (t == 0) {
        float s = b2[0];
        for (int i = 0; i < 288; ++i) s += red[i];
        out[g] = s;
    }
}

// ---------------- host driver ----------------
extern "C" void kernel_launch(void* const* d_in, const int* in_sizes, int n_in,
                              void* d_out, int out_size, void* d_ws, size_t ws_size,
                              hipStream_t stream) {
    const float* x     = (const float*)d_in[0];
    const int*   ei    = (const int*)  d_in[1];
    const float* eattr = (const float*)d_in[2];
    const int*   batch = (const int*)  d_in[3];
    const int*   ipos  = (const int*)  d_in[4];
    const float* Wf    = (const float*)d_in[5];
    const float* bfb   = (const float*)d_in[6];
    const float* Ws    = (const float*)d_in[7];
    const float* bsb   = (const float*)d_in[8];
    const float* gamma = (const float*)d_in[9];
    const float* beta  = (const float*)d_in[10];
    const float* bm    = (const float*)d_in[11];
    const float* bv    = (const float*)d_in[12];
    const float* W1    = (const float*)d_in[13];
    const float* b1    = (const float*)d_in[14];
    const float* W2    = (const float*)d_in[15];
    const float* b2    = (const float*)d_in[16];
    float* out = (float*)d_out;

    const int N  = in_sizes[0] / C_DIM;
    const int E  = in_sizes[1] / 2;
    const int Ni = in_sizes[4];
    const int G  = out_size;

    char* ws = (char*)d_ws;
    size_t off = 0;
    auto take = [&](size_t bytes) { size_t o = off; off = (off + bytes + 255) & ~(size_t)255; return o; };
    float*          h     = (float*)(ws + take((size_t)N*C_DIM*4));
    float*          agg   = (float*)(ws + take((size_t)N*C_DIM*4));
    float*          xcat  = (float*)(ws + take((size_t)N*192*4));
    float*          ea4   = (float*)(ws + take((size_t)E*4*4));
    unsigned short* Bpack = (unsigned short*)(ws + take((size_t)NLAYERS*BPACK_L_ELEMS*2));
    float*          padd  = (float*)(ws + take((size_t)G*192*4));
    unsigned*       pmax  = (unsigned*)(ws + take((size_t)G*192*4));
    float*          pcnt  = (float*)(ws + take((size_t)G*4));
    (void)ws_size; (void)n_in;

    const int* srcp = ei;
    const int* dstp = ei + E;

    pack_weights<<<(NLAYERS*FRAGS_PER_LAYER*FRAG_ELEMS + 255)/256, 256, 0, stream>>>(Wf, Ws, Bpack);
    prep_ea<<<(E + 255)/256, 256, 0, stream>>>(eattr, ea4, E);
    copy_x<<<((size_t)N*C_DIM + 255)/256, 256, 0, stream>>>(x, h, N*C_DIM);

    const int numTiles = (E + 15) / 16;
    const int eblocks  = std::min((numTiles + 3) / 4, 2500);
    for (int l = 0; l < NLAYERS; ++l) {
        fill_f32<<<((size_t)N*C_DIM + 255)/256, 256, 0, stream>>>(agg, 0.0f, N*C_DIM);
        edge_msg<<<eblocks, 128, 0, stream>>>(h, srcp, dstp, ea4,
                                              Bpack + (size_t)l*BPACK_L_ELEMS,
                                              bfb + l*C_DIM, bsb + l*C_DIM,
                                              agg, E, numTiles);
        node_update<<<((size_t)N*C_DIM + 255)/256, 256, 0, stream>>>(
            agg, h, xcat, gamma + l*C_DIM, beta + l*C_DIM,
            bm + l*C_DIM, bv + l*C_DIM, l, N*C_DIM);
    }

    fill_f32<<<(G*192 + 255)/256, 256, 0, stream>>>(padd, 0.0f, G*192);
    fill_f32<<<(G + 255)/256, 256, 0, stream>>>(pcnt, 0.0f, G);
    fill_u32<<<(G*192 + 255)/256, 256, 0, stream>>>(pmax, 0u, G*192);
    pool_scatter<<<((size_t)Ni*192 + 255)/256, 256, 0, stream>>>(xcat, ipos, batch, padd, pcnt, pmax, Ni);
    mlp_kernel<<<G, 288, 0, stream>>>(padd, pcnt, pmax, W1, b1, W2, b2, out);
}
